// SimCLR_29901562315245
// MI455X (gfx1250) — matrix operand order repaired and checked by reading, not compiled, not run on hardware
//
#include <hip/hip_runtime.h>

typedef __bf16 bf16;
typedef __attribute__((ext_vector_type(16))) __bf16 v16bf;
typedef __attribute__((ext_vector_type(8)))  float  v8f;

#define INV_T 10.0f

union FragBF { v16bf v; uint4 q[2]; };
union AccF  { v8f v; float f[8]; };
union PackBF4 { bf16 h[4]; uint2 u2; };

#define WMMA_BF16(a, b, c) \
  __builtin_amdgcn_wmma_f32_16x16x32_bf16(false, (a), false, (b), (short)0, (c), false, false)

// ---------------------------------------------------------------------------
// CDNA5 async global->LDS copy (ASYNCcnt-tracked, no VGPR data staging).
// LDS operand = low 32 bits of the generic shared-memory address (the LDS
// aperture maps addr[31:0] to the wave-relative LDS offset).
// ---------------------------------------------------------------------------
__device__ __forceinline__ void async_b128(void* lds_dst, const void* gsrc) {
  asm volatile("global_load_async_to_lds_b128 %0, %1, off"
               :: "v"((unsigned)(uintptr_t)lds_dst), "v"(gsrc)
               : "memory");
}
__device__ __forceinline__ void wait_async0() {
  asm volatile("s_wait_asynccnt 0x0" ::: "memory");
}

// ---------------------------------------------------------------------------
// WMMA fragment loaders (per ISA 7.12.2 VGPR layouts, wave32)
// A 16x32 bf16: lane l -> row (l&15); VGPR0-3 hold K = kh..kh+7 (kh=(l>>4)*8),
//               VGPR4-7 hold K = 16+kh..16+kh+7.
// B 32x16 bf16: lane l -> col (l&15); VGPR0-3 hold K = kh..kh+7 (kh=(l>>4)*8),
//               VGPR4-7 hold K = 16+kh..16+kh+7.
// ---------------------------------------------------------------------------
__device__ __forceinline__ v16bf frag_a_lds(const bf16* t, int row0, int k0, int ld) {
  const int l = threadIdx.x & 31;
  const bf16* p = t + (row0 + (l & 15)) * ld + k0 + ((l >> 4) << 3);
  FragBF f;
  f.q[0] = *reinterpret_cast<const uint4*>(p);
  f.q[1] = *reinterpret_cast<const uint4*>(p + 16);
  return f.v;
}
__device__ __forceinline__ v16bf frag_b_lds(const bf16* t, int col0, int k0, int ld) {
  const int l = threadIdx.x & 31;
  const bf16* p = t + (col0 + (l & 15)) * ld + k0 + ((l >> 4) << 3);
  FragBF f;
  f.q[0] = *reinterpret_cast<const uint4*>(p);
  f.q[1] = *reinterpret_cast<const uint4*>(p + 16);
  return f.v;
}

// ---------------------------------------------------------------------------
// Tiled GEMM: C[M,N] = act(A[M,K] @ Bt[N,K]^T + bias), bf16 WMMA, f32 accum.
// Block tile 128x128, BK=32, 256 threads (8 waves, 4x2 wave grid, each wave
// produces a 32x64 region = 2x4 fragments). Double-buffered LDS (40 KB),
// async global->LDS staging for bf16 operands.
// ---------------------------------------------------------------------------
template<bool A_F32, bool RELU, bool OUT_F32>
__global__ __launch_bounds__(256, 1)
void gemm_bias_act(const void* __restrict__ Av, const bf16* __restrict__ Bt,
                   const float* __restrict__ bias, void* __restrict__ Cv,
                   int M, int N, int K) {
  constexpr int BM = 128, BN = 128, BK = 32, LDT = BK + 8;
  __shared__ bf16 Atile[2][BM * LDT];
  __shared__ bf16 Btile[2][BN * LDT];

  const int tid = threadIdx.x;
  const int wid = tid >> 5;
  const int wm  = wid >> 1;   // 0..3 -> 32-row strip
  const int wn  = wid & 1;    // 0..1 -> 64-col strip
  const int bm  = blockIdx.y * BM;
  const int bn  = blockIdx.x * BN;

  AccF acc[2][4];
#pragma unroll
  for (int i = 0; i < 2; ++i)
#pragma unroll
    for (int j = 0; j < 4; ++j)
#pragma unroll
      for (int v = 0; v < 8; ++v) acc[i][j].f[v] = 0.f;

  auto loadA = [&](int s, int kt) {
    const int k0 = kt * BK;
    if (A_F32) {
      const float* A = (const float*)Av;
      const int c4 = tid & 7;     // 8 float4 per 32-float row
      const int r0 = tid >> 3;    // 32-row stride
#pragma unroll
      for (int i = 0; i < 4; ++i) {
        const int r = r0 + i * 32;
        float4 d = *reinterpret_cast<const float4*>(A + (size_t)(bm + r) * K + k0 + c4 * 4);
        PackBF4 pk;
        pk.h[0] = (bf16)d.x; pk.h[1] = (bf16)d.y; pk.h[2] = (bf16)d.z; pk.h[3] = (bf16)d.w;
        *reinterpret_cast<uint2*>(&Atile[s][r * LDT + c4 * 4]) = pk.u2;
      }
    } else {
      const bf16* A = (const bf16*)Av;
      const int c8 = tid & 3;     // 4 x b128 per 32-bf16 row
      const int r0 = tid >> 2;    // 64-row stride
#pragma unroll
      for (int i = 0; i < 2; ++i) {
        const int r = r0 + i * 64;
        async_b128(&Atile[s][r * LDT + c8 * 8], A + (size_t)(bm + r) * K + k0 + c8 * 8);
      }
    }
  };
  auto loadB = [&](int s, int kt) {
    const int k0 = kt * BK;
    const int c8 = tid & 3;
    const int r0 = tid >> 2;
#pragma unroll
    for (int i = 0; i < 2; ++i) {
      const int r = r0 + i * 64;
      async_b128(&Btile[s][r * LDT + c8 * 8], Bt + (size_t)(bn + r) * K + k0 + c8 * 8);
    }
  };

  const int KT = K / BK;
  loadA(0, 0); loadB(0, 0);
  wait_async0();
  __syncthreads();

#pragma unroll 1
  for (int kt = 0; kt < KT; ++kt) {
    const int cur = kt & 1;
    if (kt + 1 < KT) { loadA(cur ^ 1, kt + 1); loadB(cur ^ 1, kt + 1); }

    v16bf a0 = frag_a_lds(Atile[cur], wm * 32,      0, LDT);
    v16bf a1 = frag_a_lds(Atile[cur], wm * 32 + 16, 0, LDT);
    v16bf b0 = frag_b_lds(Btile[cur], wn * 64,      0, LDT);
    v16bf b1 = frag_b_lds(Btile[cur], wn * 64 + 16, 0, LDT);
    v16bf b2 = frag_b_lds(Btile[cur], wn * 64 + 32, 0, LDT);
    v16bf b3 = frag_b_lds(Btile[cur], wn * 64 + 48, 0, LDT);

    acc[0][0].v = WMMA_BF16(a0, b0, acc[0][0].v);
    acc[0][1].v = WMMA_BF16(a0, b1, acc[0][1].v);
    acc[0][2].v = WMMA_BF16(a0, b2, acc[0][2].v);
    acc[0][3].v = WMMA_BF16(a0, b3, acc[0][3].v);
    acc[1][0].v = WMMA_BF16(a1, b0, acc[1][0].v);
    acc[1][1].v = WMMA_BF16(a1, b1, acc[1][1].v);
    acc[1][2].v = WMMA_BF16(a1, b2, acc[1][2].v);
    acc[1][3].v = WMMA_BF16(a1, b3, acc[1][3].v);

    wait_async0();
    __syncthreads();
  }

  // Epilogue: C/D layout -> VGPR v, lane l: row = v + 8*(l>=16), col = l&15
  const int l    = tid & 31;
  const int lrow = (l >> 4) << 3;
  const int lcol = l & 15;
#pragma unroll
  for (int i = 0; i < 2; ++i) {
#pragma unroll
    for (int j = 0; j < 4; ++j) {
      const int col = bn + wn * 64 + j * 16 + lcol;
      const float bv = bias[col];
#pragma unroll
      for (int v = 0; v < 8; ++v) {
        const int row = bm + wm * 32 + i * 16 + lrow + v;
        float x = acc[i][j].f[v] + bv;
        if (RELU) x = fmaxf(x, 0.f);
        if (OUT_F32) ((float*)Cv)[(size_t)row * N + col] = x;
        else         ((bf16*)Cv)[(size_t)row * N + col] = (bf16)x;
      }
    }
  }
}

// ---------------------------------------------------------------------------
// Weight convert + transpose: W[K][N] f32 -> Wt[N][K] bf16 (one-time cost)
// ---------------------------------------------------------------------------
__global__ void convert_transpose(const float* __restrict__ W, bf16* __restrict__ Wt,
                                  int K, int N) {
  size_t idx = (size_t)blockIdx.x * blockDim.x + threadIdx.x;
  size_t total = (size_t)K * N;
  if (idx >= total) return;
  size_t n = idx / K, k = idx % K;
  Wt[idx] = (bf16)W[k * N + n];
}

__global__ void zero_f32(float* __restrict__ p, int n) {
  int i = blockIdx.x * blockDim.x + threadIdx.x;
  if (i < n) p[i] = 0.f;
}

// One wave per row (128 cols): sum of squares via shuffle, write bf16 row.
__global__ __launch_bounds__(256)
void normalize_rows(const float* __restrict__ P, bf16* __restrict__ R) {
  const int l   = threadIdx.x & 31;
  const int row = blockIdx.x * 8 + (threadIdx.x >> 5);
  float4 d = *reinterpret_cast<const float4*>(P + (size_t)row * 128 + l * 4);
  float s = d.x * d.x + d.y * d.y + d.z * d.z + d.w * d.w;
  s += __shfl_xor(s, 16); s += __shfl_xor(s, 8); s += __shfl_xor(s, 4);
  s += __shfl_xor(s, 2);  s += __shfl_xor(s, 1);
  const float inv = 1.0f / fmaxf(sqrtf(s), 1e-12f);
  bf16* r = R + (size_t)row * 128 + l * 4;
  r[0] = (bf16)(d.x * inv); r[1] = (bf16)(d.y * inv);
  r[2] = (bf16)(d.z * inv); r[3] = (bf16)(d.w * inv);
}

// ---------------------------------------------------------------------------
// Similarity + streaming logsumexp. Block = 128 rows of r1; A fragments for
// all K=128 kept in registers; r2 column tiles (128x128 bf16) staged in LDS
// via async copies. exp shift fixed at 1/T (cosine sims in [-1,1] => no
// overflow). Diagonal of S is pos_dot; per-row sums combined via f32 atomics.
// ---------------------------------------------------------------------------
__global__ __launch_bounds__(256, 1)
void sim_lse(const bf16* __restrict__ R1, const bf16* __restrict__ R2,
             float* __restrict__ pos, float* __restrict__ denom, int Nrows) {
  constexpr int BT = 128, KD = 128, LDT = KD + 8;
  __shared__ bf16 T2[BT * LDT];   // 34.8 KB

  const int tid = threadIdx.x;
  const int wid = tid >> 5;
  const int wm  = wid >> 1;
  const int wn  = wid & 1;
  const int l   = tid & 31;
  const int bm  = blockIdx.x * BT;

  // r1 fragments for this wave's 32 rows, all 4 K-slices, loaded once.
  v16bf afr[2][4];
#pragma unroll
  for (int i = 0; i < 2; ++i)
#pragma unroll
    for (int ks = 0; ks < 4; ++ks) {
      const bf16* p = R1 + (size_t)(bm + wm * 32 + i * 16 + (l & 15)) * KD
                         + ks * 32 + ((l >> 4) << 3);
      FragBF f;
      f.q[0] = *reinterpret_cast<const uint4*>(p);
      f.q[1] = *reinterpret_cast<const uint4*>(p + 16);
      afr[i][ks] = f.v;
    }

  float ssum[16];
#pragma unroll
  for (int i = 0; i < 16; ++i) ssum[i] = 0.f;

  const int lrow = (l >> 4) << 3;
  const int lcol = l & 15;
  const int CT = Nrows / BT;

#pragma unroll 1
  for (int ct = 0; ct < CT; ++ct) {
    // stage r2 column tile (async; previous iteration's barrier guarantees
    // no wave is still reading T2)
    {
      const int c  = tid & 15;     // 16 x b128 per 128-bf16 row
      const int r0 = tid >> 4;     // 16-row stride
#pragma unroll
      for (int i = 0; i < 8; ++i) {
        const int r = r0 + i * 16;
        async_b128(&T2[r * LDT + c * 8], R2 + (size_t)(ct * BT + r) * KD + c * 8);
      }
    }
    wait_async0();
    __syncthreads();

    AccF acc[2][4];
#pragma unroll
    for (int i = 0; i < 2; ++i)
#pragma unroll
      for (int j = 0; j < 4; ++j)
#pragma unroll
        for (int v = 0; v < 8; ++v) acc[i][j].f[v] = 0.f;

#pragma unroll
    for (int ks = 0; ks < 4; ++ks) {
      const int k0 = ks * 32;
      // per-j structure: one live B fragment, two WMMAs on it
#pragma unroll
      for (int j = 0; j < 4; ++j) {
        v16bf b = frag_b_lds(T2, wn * 64 + j * 16, k0, LDT);
        acc[0][j].v = WMMA_BF16(afr[0][ks], b, acc[0][j].v);
        acc[1][j].v = WMMA_BF16(afr[1][ks], b, acc[1][j].v);
      }
    }

    // fused epilogue: scale by 1/T, capture diagonal, accumulate exp(x - 1/T)
#pragma unroll
    for (int i = 0; i < 2; ++i)
#pragma unroll
      for (int j = 0; j < 4; ++j) {
        const int col = ct * BT + wn * 64 + j * 16 + lcol;
#pragma unroll
        for (int v = 0; v < 8; ++v) {
          const float x = acc[i][j].f[v] * INV_T;
          const int row = bm + wm * 32 + i * 16 + lrow + v;
          if (col == row) pos[row] = x;
          ssum[i * 8 + v] += __expf(x - INV_T);
        }
      }
    __syncthreads();
  }

  // reduce partial sums across the 16 lanes holding the same rows
#pragma unroll
  for (int i = 0; i < 16; ++i) {
    float s = ssum[i];
    s += __shfl_xor(s, 1); s += __shfl_xor(s, 2);
    s += __shfl_xor(s, 4); s += __shfl_xor(s, 8);
    ssum[i] = s;
  }
  if ((l & 15) == 0) {
#pragma unroll
    for (int i = 0; i < 16; ++i) {
      const int ti = i >> 3, v = i & 7;
      const int row = bm + wm * 32 + ti * 16 + lrow + v;
      atomicAdd(&denom[row], ssum[i]);  // combines the two wn column halves
    }
  }
}

__global__ __launch_bounds__(256)
void loss_reduce(const float* __restrict__ pos, const float* __restrict__ denom,
                 float* __restrict__ out, int n) {
  __shared__ float sm[256];
  float s = 0.f;
  for (int i = threadIdx.x; i < n; i += 256)
    s += -pos[i] + (INV_T + __logf(denom[i]));
  sm[threadIdx.x] = s;
  __syncthreads();
  for (int st = 128; st > 0; st >>= 1) {
    if ((int)threadIdx.x < st) sm[threadIdx.x] += sm[threadIdx.x + st];
    __syncthreads();
  }
  if (threadIdx.x == 0) out[0] = sm[0] / (float)n;
}

// ---------------------------------------------------------------------------
extern "C" void kernel_launch(void* const* d_in, const int* in_sizes, int n_in,
                              void* d_out, int out_size, void* d_ws, size_t ws_size,
                              hipStream_t stream) {
  const float* f1 = (const float*)d_in[0];
  const float* f2 = (const float*)d_in[1];
  const float* W1 = (const float*)d_in[2];
  const float* b1 = (const float*)d_in[3];
  const float* W2 = (const float*)d_in[4];
  const float* b2 = (const float*)d_in[5];
  const float* W3 = (const float*)d_in[6];
  const float* b3 = (const float*)d_in[7];
  float* out = (float*)d_out;

  const int N = 8192, F = 1024, H = 2048, P = 128;

  char* ws = (char*)d_ws;
  size_t off = 0;
  auto alloc = [&](size_t bytes) -> void* {
    void* p = ws + off;
    off += (bytes + 255) & ~(size_t)255;
    return p;
  };
  bf16*  W1t  = (bf16*)alloc((size_t)H * F * 2);
  bf16*  W2t  = (bf16*)alloc((size_t)H * H * 2);
  bf16*  W3t  = (bf16*)alloc((size_t)P * H * 2);
  bf16*  H1   = (bf16*)alloc((size_t)N * H * 2);
  bf16*  H2   = (bf16*)alloc((size_t)N * H * 2);
  float* proj = (float*)alloc((size_t)N * P * 4);
  bf16*  r1   = (bf16*)alloc((size_t)N * P * 2);
  bf16*  r2   = (bf16*)alloc((size_t)N * P * 2);
  float* pos  = (float*)alloc((size_t)N * 4);
  float* den  = (float*)alloc((size_t)N * 4);

  convert_transpose<<<(int)(((size_t)H * F + 255) / 256), 256, 0, stream>>>(W1, W1t, F, H);
  convert_transpose<<<(int)(((size_t)H * H + 255) / 256), 256, 0, stream>>>(W2, W2t, H, H);
  convert_transpose<<<(int)(((size_t)P * H + 255) / 256), 256, 0, stream>>>(W3, W3t, H, P);
  zero_f32<<<(N + 255) / 256, 256, 0, stream>>>(den, N);
  zero_f32<<<(N + 255) / 256, 256, 0, stream>>>(pos, N);

  for (int v = 0; v < 2; ++v) {
    const float* fv = v ? f2 : f1;
    bf16* rv = v ? r2 : r1;
    gemm_bias_act<true,  true,  false><<<dim3(H / 128, N / 128), 256, 0, stream>>>(fv, W1t, b1, H1, N, H, F);
    gemm_bias_act<false, true,  false><<<dim3(H / 128, N / 128), 256, 0, stream>>>(H1, W2t, b2, H2, N, H, H);
    gemm_bias_act<false, false, true ><<<dim3(P / 128, N / 128), 256, 0, stream>>>(H2, W3t, b3, proj, N, P, H);
    normalize_rows<<<N / 8, 256, 0, stream>>>(proj, rv);
  }

  sim_lse<<<N / 128, 256, 0, stream>>>(r1, r2, pos, den, N);
  loss_reduce<<<1, 256, 0, stream>>>(pos, den, out, N);
}
